// Net_13219909337298
// MI455X (gfx1250) — matrix operand order repaired and checked by reading, not compiled, not run on hardware
//
#include <hip/hip_runtime.h>
#include <hip/hip_bf16.h>

// ---- problem constants (match reference) ----
#define T_STEPS 2048
#define IO_W    32
#define A_W     16
#define O_W     8
#define H_W     1024
#define H3      3072
#define DEPTH   5

typedef __attribute__((ext_vector_type(16))) __bf16 v16bf;
typedef __attribute__((ext_vector_type(8)))  __bf16 v8bf;
typedef __attribute__((ext_vector_type(8)))  float  v8f;

union V16U { v16bf v; v8bf h[2]; };

// A fragment: 16x32 bf16 tile, row-major source. Lane m=lane&15 holds row m.
// lanes 0-15: K = {0..7, 16..23}; lanes 16-31: K = {8..15, 24..31}  (ISA 7.12.2)
__device__ __forceinline__ v16bf load_a_tile(const __bf16* base, int ld, int lane) {
  int row = lane & 15;
  int klo = (lane < 16) ? 0 : 8;
  const __bf16* p = base + (size_t)row * ld;
  V16U u;
  u.h[0] = *(const v8bf*)(p + klo);
  u.h[1] = *(const v8bf*)(p + klo + 16);
  return u.v;
}

// Same, but non-temporal: used for streamed weights that cannot fit in WGP$/L0,
// so they should not evict the reused x/h broadcast lines.
__device__ __forceinline__ v16bf load_a_tile_nt(const __bf16* base, int ld, int lane) {
  int row = lane & 15;
  int klo = (lane < 16) ? 0 : 8;
  const __bf16* p = base + (size_t)row * ld;
  V16U u;
  u.h[0] = __builtin_nontemporal_load((const v8bf*)(p + klo));
  u.h[1] = __builtin_nontemporal_load((const v8bf*)(p + klo + 16));
  return u.v;
}

// B fragment: 32x16 bf16 (we feed W row-major as B = W^T). Lane n=lane&15 holds col n.
// lanes 0-15: K = {0..7, 16..23}; lanes 16-31: K = {8..15, 24..31} (matches A)
__device__ __forceinline__ v16bf load_b_tile(const __bf16* base, int ld, int lane) {
  int col = lane & 15;
  int kb  = (lane < 16) ? 0 : 8;
  const __bf16* p = base + (size_t)col * ld + kb;
  V16U u;
  u.h[0] = *(const v8bf*)p;
  u.h[1] = *(const v8bf*)(p + 16);
  return u.v;
}

// B fragment holding the same x-vector segment in every column (matvec use):
// every output column of D then equals W_tile * x_seg.
__device__ __forceinline__ v16bf load_x_bcast(const __bf16* x, int lane) {
  int kb = (lane < 16) ? 0 : 8;
  V16U u;
  u.h[0] = *(const v8bf*)(x + kb);
  u.h[1] = *(const v8bf*)(x + kb + 16);
  return u.v;
}

// ---------------- parallel GEMM: out[M,N] = A[M,K] @ B[N,K]^T (+bias) -------------
// One wave computes a 16x32 output tile (A fragment reused across two WMMAs).
// mode 0: store bf16 [M,N]; mode 1: store f32 [M,N]; mode 2: store f32 to d_out as
// [O, T, IO] with o=n/32, io=n%32, t=m.
__global__ __launch_bounds__(128) void wmma_gemm(
    const __bf16* __restrict__ A, const __bf16* __restrict__ B,
    const float* __restrict__ bias, void* __restrict__ out,
    int M, int N, int K, int mode) {
  int lane = threadIdx.x & 31;
  int gw = (blockIdx.x * blockDim.x + threadIdx.x) >> 5;
  int tn = N >> 5;                    // 32-wide N tiles
  int tiles = (M >> 4) * tn;
  if (gw >= tiles) return;
  int m0 = (gw / tn) << 4;
  int n0 = (gw % tn) << 5;
  const __bf16* Ap = A + (size_t)m0 * K;
  const __bf16* B0 = B + (size_t)n0 * K;
  const __bf16* B1 = B0 + (size_t)16 * K;
  v8f acc0 = {}, acc1 = {};
#pragma unroll 4
  for (int k = 0; k < K; k += 32) {
    v16bf a  = load_a_tile(Ap + k, K, lane);
    v16bf b0 = load_b_tile(B0 + k, K, lane);
    v16bf b1 = load_b_tile(B1 + k, K, lane);
    acc0 = __builtin_amdgcn_wmma_f32_16x16x32_bf16(false, a, false, b0, (short)0, acc0,
                                                   false, false);
    acc1 = __builtin_amdgcn_wmma_f32_16x16x32_bf16(false, a, false, b1, (short)0, acc1,
                                                   false, false);
  }
  int mbase = m0 + ((lane < 16) ? 0 : 8);
#pragma unroll
  for (int j = 0; j < 2; ++j) {
    int n = n0 + j * 16 + (lane & 15);
    float bv = bias ? bias[n] : 0.0f;
    const v8f& acc = j ? acc1 : acc0;
#pragma unroll
    for (int r = 0; r < 8; ++r) {
      float v = acc[r] + bv;
      int m = mbase + r;
      if (mode == 0)
        ((__bf16*)out)[(size_t)m * N + n] = (__bf16)v;
      else if (mode == 1)
        ((float*)out)[(size_t)m * N + n] = v;
      else
        ((float*)out)[(size_t)(n >> 5) * (T_STEPS * IO_W) + (size_t)m * IO_W + (n & 31)] = v;
    }
  }
}

// ---------------- persistent sequential GRU (layer-pipelined wavefront) ----------
// 9 matvec units (gh l=0..4, gi l=1..4), each 3072 rows = 96 wave-tiles of 32 rows.
#define SEQ_BLOCKS 108   // 9 * 96 waves / 8 waves per block

__device__ __forceinline__ void gbar(int* cnt, int* sense, int pid) {
  __threadfence();          // make this thread's global writes visible
  __syncthreads();          // all waves in block have fenced
  if (threadIdx.x == 0) {
    if (__hip_atomic_fetch_add(cnt, 1, __ATOMIC_ACQ_REL, __HIP_MEMORY_SCOPE_AGENT)
        == SEQ_BLOCKS - 1) {
      __hip_atomic_store(cnt, 0, __ATOMIC_RELAXED, __HIP_MEMORY_SCOPE_AGENT);
      __hip_atomic_store(sense, pid + 1, __ATOMIC_RELEASE, __HIP_MEMORY_SCOPE_AGENT);
    } else {
      while (__hip_atomic_load(sense, __ATOMIC_ACQUIRE, __HIP_MEMORY_SCOPE_AGENT) <= pid)
        __builtin_amdgcn_s_sleep(1);
    }
  }
  __syncthreads();
}

// Units: 0..4 = gh-matvec of layer u (Whh_l @ h_l(t-1));
//        5..8 = gi-matvec of layer u-4 (Wih_l @ h_{l-1}(t)), layer 0 gi precomputed.
__global__ __launch_bounds__(256) void gru_seq(
    const __bf16* __restrict__ wih, const __bf16* __restrict__ whh,
    const float* __restrict__ bih, const float* __restrict__ bhh,
    const float* __restrict__ Gi0, float* __restrict__ Gi, float* __restrict__ Gh,
    float* __restrict__ H, __bf16* __restrict__ Hbf, __bf16* __restrict__ gout,
    int* __restrict__ bar) {
  int lane = threadIdx.x & 31;
  int gw = blockIdx.x * 8 + (threadIdx.x >> 5);
  int unit = gw / 96;           // 0..8
  int row0 = (gw % 96) << 5;    // 32-row tile within the 3072-row matvec
  int* cnt = bar;
  int* sense = bar + 1;
  int pid = 0;

  for (int s = 0; s < T_STEPS + DEPTH - 1; ++s) {
    // ------- matvec phase: reads h state written at step s-1 -------
    {
      bool is_gh = (unit < DEPTH);
      int l = is_gh ? unit : (unit - DEPTH + 1);   // gi units: l = 1..4
      int t = s - l;
      if (t >= 0 && t < T_STEPS) {
        const __bf16* W0 =
            (is_gh ? whh : wih) + (size_t)l * H3 * H_W + (size_t)row0 * H_W;
        const __bf16* W1 = W0 + (size_t)16 * H_W;
        const __bf16* x = is_gh ? (Hbf + (size_t)(l * 2 + ((t - 1) & 1)) * H_W)
                                : (Hbf + (size_t)((l - 1) * 2 + (t & 1)) * H_W);
        v8f acc0 = {}, acc1 = {};
#pragma unroll 4
        for (int k = 0; k < H_W; k += 32) {
          v16bf b  = load_x_bcast(x + k, lane);
          v16bf a0 = load_a_tile_nt(W0 + k, H_W, lane);
          v16bf a1 = load_a_tile_nt(W1 + k, H_W, lane);
          acc0 = __builtin_amdgcn_wmma_f32_16x16x32_bf16(false, a0, false, b, (short)0,
                                                         acc0, false, false);
          acc1 = __builtin_amdgcn_wmma_f32_16x16x32_bf16(false, a1, false, b, (short)0,
                                                         acc1, false, false);
        }
        // column 0 of D holds y: lane0 -> rows +0..7, lane16 -> rows +8..15
        float* dst = (is_gh ? Gh : Gi) + l * H3 + row0;
        if (lane == 0) {
#pragma unroll
          for (int r = 0; r < 8; ++r) { dst[r] = acc0[r]; dst[16 + r] = acc1[r]; }
        } else if (lane == 16) {
#pragma unroll
          for (int r = 0; r < 8; ++r) { dst[8 + r] = acc0[r]; dst[24 + r] = acc1[r]; }
        }
      }
    }
    gbar(cnt, sense, pid++);

    // ------- elementwise GRU cell phase (5*1024 elems, first 20 blocks) -------
    {
      int gid = blockIdx.x * 256 + threadIdx.x;
      if (gid < DEPTH * H_W) {
        int le = gid >> 10, i = gid & (H_W - 1);
        int te = s - le;
        if (te >= 0 && te < T_STEPS) {
          float gir, giz, gin;
          if (le == 0) {
            const float* g = Gi0 + (size_t)te * H3;
            gir = g[i]; giz = g[H_W + i]; gin = g[2 * H_W + i];
          } else {
            const float* g = Gi + le * H3;
            gir = g[i]; giz = g[H_W + i]; gin = g[2 * H_W + i];
          }
          const float* gh = Gh + le * H3;
          const float* bi = bih + le * H3;
          const float* bh = bhh + le * H3;
          float hprev = H[(size_t)(le * 2 + ((te - 1) & 1)) * H_W + i];
          float r = 1.0f / (1.0f + __expf(-(gir + bi[i] + gh[i] + bh[i])));
          float z = 1.0f / (1.0f + __expf(-(giz + bi[H_W + i] + gh[H_W + i] + bh[H_W + i])));
          float n = tanhf(gin + bi[2 * H_W + i] + r * (gh[2 * H_W + i] + bh[2 * H_W + i]));
          float hn = (1.0f - z) * n + z * hprev;
          size_t slot = (size_t)(le * 2 + (te & 1)) * H_W + i;
          H[slot] = hn;
          Hbf[slot] = (__bf16)hn;
          if (le == DEPTH - 1) gout[(size_t)te * H_W + i] = (__bf16)hn;
        }
      }
    }
    gbar(cnt, sense, pid++);
  }
}

// ---------------- prep kernels ----------------
__global__ void k_convert(const float* __restrict__ s, __bf16* __restrict__ d, int n) {
  int i = blockIdx.x * blockDim.x + threadIdx.x;
  if (i < n) d[i] = (__bf16)s[i];
}

// lines[t, f*32+c] = f==0 ? x[t,c] : x[t-(f+1), c] (zero if t-(f+1) < 0)
__global__ void k_lines(const float* __restrict__ x, __bf16* __restrict__ lines) {
  int i = blockIdx.x * blockDim.x + threadIdx.x;
  if (i < T_STEPS * IO_W * A_W) {
    int t = i >> 9, j = i & 511;
    int f = j >> 5, c = j & 31;
    int tt = (f == 0) ? t : (t - (f + 1));
    lines[i] = (__bf16)((tt >= 0) ? x[tt * IO_W + c] : 0.0f);
  }
}

__global__ void k_init(float* __restrict__ H, __bf16* __restrict__ Hbf,
                       int* __restrict__ bar) {
  int i = blockIdx.x * blockDim.x + threadIdx.x;
  if (i < DEPTH * 2 * H_W) { H[i] = 0.0f; Hbf[i] = (__bf16)0.0f; }
  if (i < 2) bar[i] = 0;
}

// ---------------- host side ----------------
static inline size_t algn256(size_t x) { return (x + 255) & ~(size_t)255; }

extern "C" void kernel_launch(void* const* d_in, const int* in_sizes, int n_in,
                              void* d_out, int out_size, void* d_ws, size_t ws_size,
                              hipStream_t stream) {
  (void)in_sizes; (void)n_in; (void)out_size; (void)ws_size;
  const float* x    = (const float*)d_in[0];
  const float* w1   = (const float*)d_in[1];
  const float* b1   = (const float*)d_in[2];
  const float* wih  = (const float*)d_in[3];
  const float* whh  = (const float*)d_in[4];
  const float* bih  = (const float*)d_in[5];
  const float* bhh  = (const float*)d_in[6];
  const float* w2   = (const float*)d_in[7];
  const float* b2   = (const float*)d_in[8];
  float* out = (float*)d_out;

  char* ws = (char*)d_ws;
  size_t off = 0;
  auto take = [&](size_t bytes) -> void* {
    void* p = ws + off;
    off = algn256(off + bytes);
    return p;
  };
  int*    bar     = (int*)   take(256);
  __bf16* w1bf    = (__bf16*)take((size_t)H_W * (IO_W * A_W) * 2);        // 1 MB
  __bf16* wihbf   = (__bf16*)take((size_t)DEPTH * H3 * H_W * 2);         // 31.5 MB
  __bf16* whhbf   = (__bf16*)take((size_t)DEPTH * H3 * H_W * 2);         // 31.5 MB
  __bf16* w2bf    = (__bf16*)take((size_t)(IO_W * O_W) * H_W * 2);       // 0.5 MB
  __bf16* linesbf = (__bf16*)take((size_t)T_STEPS * (IO_W * A_W) * 2);   // 2 MB
  __bf16* prebf   = (__bf16*)take((size_t)T_STEPS * H_W * 2);            // 4 MB
  float*  Gi0     = (float*) take((size_t)T_STEPS * H3 * 4);             // 25 MB
  float*  Gi      = (float*) take((size_t)DEPTH * H3 * 4);
  float*  Gh      = (float*) take((size_t)DEPTH * H3 * 4);
  float*  Hst     = (float*) take((size_t)DEPTH * 2 * H_W * 4);
  __bf16* Hbf     = (__bf16*)take((size_t)DEPTH * 2 * H_W * 2);
  __bf16* goutbf  = (__bf16*)take((size_t)T_STEPS * H_W * 2);            // 4 MB

  // init state + barrier
  k_init<<<40, 256, 0, stream>>>(Hst, Hbf, bar);

  // fp32 -> bf16 weight conversion (weights then stay L2-resident, 63 MB total)
  int nw1 = H_W * IO_W * A_W;
  int nwg = DEPTH * H3 * H_W;
  int nw2 = IO_W * O_W * H_W;
  k_convert<<<(nw1 + 255) / 256, 256, 0, stream>>>(w1, w1bf, nw1);
  k_convert<<<(nwg + 255) / 256, 256, 0, stream>>>(wih, wihbf, nwg);
  k_convert<<<(nwg + 255) / 256, 256, 0, stream>>>(whh, whhbf, nwg);
  k_convert<<<(nw2 + 255) / 256, 256, 0, stream>>>(w2, w2bf, nw2);
  k_lines<<<(T_STEPS * IO_W * A_W + 255) / 256, 256, 0, stream>>>(x, linesbf);

  // lin1: pre[T,H] = lines @ w1^T + b1   (M=2048, N=1024, K=512) -> bf16
  {
    int tiles = (T_STEPS / 16) * (H_W / 32);
    wmma_gemm<<<(tiles * 32 + 127) / 128, 128, 0, stream>>>(
        linesbf, w1bf, b1, prebf, T_STEPS, H_W, IO_W * A_W, 0);
  }
  // Gi0[T,3H] = pre @ wih0^T (bias added in cell phase)  (M=2048, N=3072, K=1024)
  {
    int tiles = (T_STEPS / 16) * (H3 / 32);
    wmma_gemm<<<(tiles * 32 + 127) / 128, 128, 0, stream>>>(
        prebf, wihbf, (const float*)nullptr, Gi0, T_STEPS, H3, H_W, 1);
  }
  // sequential GRU: persistent kernel, layer-pipelined wavefront
  gru_seq<<<SEQ_BLOCKS, 256, 0, stream>>>(wihbf, whhbf, bih, bhh, Gi0, Gi, Gh,
                                          Hst, Hbf, goutbf, bar);
  // lin2 + transpose into d_out [O, T, IO]  (M=2048, N=256, K=1024)
  {
    int tiles = (T_STEPS / 16) * ((IO_W * O_W) / 32);
    wmma_gemm<<<(tiles * 32 + 127) / 128, 128, 0, stream>>>(
        goutbf, w2bf, b2, out, T_STEPS, IO_W * O_W, H_W, 2);
  }
}
